// GCN_37890201485969
// MI455X (gfx1250) — matrix operand order repaired and checked by reading, not compiled, not run on hardware
//
#include <hip/hip_runtime.h>

// ---------------------------------------------------------------------------
// GCN forward for MI455X (gfx1250, wave32).
// GEMMs via v_wmma_f32_16x16x32_f16 (weights in LDS f16, padded rows ->
// conflict-free ds_load_b128; B double-buffered against the wmma stream).
// SpMM: CSR-by-destination built once per launch (int-atomic count + scan +
// bucket fill), then wave-per-row gather aggregation -> NO float atomics.
// LN / bias / ReLU / residual fused into epilogues.
// ---------------------------------------------------------------------------

typedef _Float16 v16h __attribute__((ext_vector_type(16)));
typedef _Float16 v8h  __attribute__((ext_vector_type(8)));
typedef float    v8f  __attribute__((ext_vector_type(8)));
typedef float    v4f  __attribute__((ext_vector_type(4)));

constexpr int   NNODE  = 100000;
constexpr int   NEDGE  = 3200000;
constexpr int   NFEAT  = 500;
constexpr int   NHIDC  = 256;
constexpr int   NCLS   = 40;
constexpr int   NLAY   = 5;
constexpr float EPSLN  = 1e-5f;

constexpr int LDS_CONV_STRIDE = NHIDC + 8;  // 264 halves: 132 dwords/row -> rows 4 banks apart
constexpr int LDS_IN_STRIDE   = 512 + 8;    // 520 halves: 260 dwords/row -> rows 4 banks apart

// ---------------- wave helpers (wave32) ----------------
__device__ __forceinline__ float wave_sum32(float v) {
#pragma unroll
    for (int off = 16; off > 0; off >>= 1) v += __shfl_xor(v, off, 32);
    return v;
}

// A fragment: 16x32 f16, M rows across lanes (lane&15), K split per ISA 7.12.2:
// lane<16: elems 0..7 = K0+0..7, elems 8..15 = K0+16..23
// lane>=16: elems 0..7 = K0+8..15, elems 8..15 = K0+24..31
__device__ __forceinline__ v16h load_a_frag(const float* __restrict__ Arow, int K0, int lane) {
    const int koff = (lane < 16) ? 0 : 8;
    const v4f* p0 = (const v4f*)(Arow + K0 + koff);       // 32B-aligned runs
    const v4f* p1 = (const v4f*)(Arow + K0 + 16 + koff);
    v4f x0 = p0[0], x1 = p0[1];
    v4f y0 = p1[0], y1 = p1[1];
    v16h a;
#pragma unroll
    for (int e = 0; e < 4; e++) {
        a[e]      = (_Float16)x0[e];
        a[4 + e]  = (_Float16)x1[e];
        a[8 + e]  = (_Float16)y0[e];
        a[12 + e] = (_Float16)y1[e];
    }
    return a;
}

__device__ __forceinline__ v16h load_a_frag_guard(const float* __restrict__ Arow, int K0,
                                                  int lane, int kmax) {
    const int koff = (lane < 16) ? 0 : 8;
    v16h a;
#pragma unroll
    for (int e = 0; e < 8; e++) {
        int k0 = K0 + koff + e;
        int k1 = K0 + 16 + koff + e;
        a[e]     = (_Float16)((k0 < kmax) ? Arow[k0] : 0.0f);
        a[8 + e] = (_Float16)((k1 < kmax) ? Arow[k1] : 0.0f);
    }
    return a;
}

// B fragment: 32x16 f16, B[k][n] = W[n][k]. N across lanes (lane&15),
// lane<16: K0+0..7, K0+16..23; lane>=16: K0+8..15, K0+24..31 (same split as A).
__device__ __forceinline__ v16h load_b_frag(const _Float16* __restrict__ sW, int stride,
                                            int n0, int K0, int lane) {
    const int n  = n0 + (lane & 15);
    const int kb = K0 + ((lane < 16) ? 0 : 8);
    const v8h* p = (const v8h*)(sW + n * stride + kb);    // 16B aligned
    v8h b0 = p[0], b1 = p[2];
    v16h b;
#pragma unroll
    for (int e = 0; e < 8; e++) { b[e] = b0[e]; b[8 + e] = b1[e]; }
    return b;
}

__device__ __forceinline__ v8f wmma_f16(v16h a, v16h b, v8f c) {
    return __builtin_amdgcn_wmma_f32_16x16x32_f16(false, a, false, b, (short)0, c,
                                                  false, false);
}

// ---------------- LayerNorm: one wave per row ----------------
__global__ __launch_bounds__(256) void ln_kernel(const float* __restrict__ h,
                                                 const float* __restrict__ g,
                                                 const float* __restrict__ b,
                                                 float* __restrict__ t, int nrows) {
    const int wid  = (blockIdx.x * blockDim.x + threadIdx.x) >> 5;
    const int lane = threadIdx.x & 31;
    if (wid >= nrows) return;
    const float4* hp = (const float4*)(h + (size_t)wid * NHIDC);
    float4 a0 = hp[lane * 2], a1 = hp[lane * 2 + 1];
    float s  = a0.x + a0.y + a0.z + a0.w + a1.x + a1.y + a1.z + a1.w;
    float s2 = a0.x*a0.x + a0.y*a0.y + a0.z*a0.z + a0.w*a0.w
             + a1.x*a1.x + a1.y*a1.y + a1.z*a1.z + a1.w*a1.w;
    s  = wave_sum32(s);
    s2 = wave_sum32(s2);
    const float mu  = s * (1.0f / NHIDC);
    const float var = s2 * (1.0f / NHIDC) - mu * mu;
    const float rs  = rsqrtf(var + EPSLN);
    const float4* gp = (const float4*)g;
    const float4* bp = (const float4*)b;
    float4 g0 = gp[lane * 2], g1 = gp[lane * 2 + 1];
    float4 b0 = bp[lane * 2], b1 = bp[lane * 2 + 1];
    float4 o0, o1;
    o0.x = (a0.x - mu) * rs * g0.x + b0.x;  o0.y = (a0.y - mu) * rs * g0.y + b0.y;
    o0.z = (a0.z - mu) * rs * g0.z + b0.z;  o0.w = (a0.w - mu) * rs * g0.w + b0.w;
    o1.x = (a1.x - mu) * rs * g1.x + b1.x;  o1.y = (a1.y - mu) * rs * g1.y + b1.y;
    o1.z = (a1.z - mu) * rs * g1.z + b1.z;  o1.w = (a1.w - mu) * rs * g1.w + b1.w;
    float4* tp = (float4*)(t + (size_t)wid * NHIDC);
    tp[lane * 2] = o0;
    tp[lane * 2 + 1] = o1;
}

// ---------------- CSR build: counts -> exclusive scan -> bucket fill ----------------
__global__ __launch_bounds__(256) void count_kernel(const int* __restrict__ row,
                                                    int* __restrict__ counts, int nedges) {
    int e = blockIdx.x * 256 + threadIdx.x;
    if (e < nedges) atomicAdd(counts + row[e], 1);
}

// 1024 elements per block: per-thread serial prefix of 4 + Hillis-Steele over 256 sums.
__global__ __launch_bounds__(256) void scan1_kernel(const int* __restrict__ counts,
                                                    int* __restrict__ offs,
                                                    int* __restrict__ blockSums, int n) {
    __shared__ int lds[256];
    const int tid = threadIdx.x;
    const int base = blockIdx.x * 1024 + tid * 4;
    int c0 = (base + 0 < n) ? counts[base + 0] : 0;
    int c1 = (base + 1 < n) ? counts[base + 1] : 0;
    int c2 = (base + 2 < n) ? counts[base + 2] : 0;
    int c3 = (base + 3 < n) ? counts[base + 3] : 0;
    const int tsum = c0 + c1 + c2 + c3;
    lds[tid] = tsum;
    __syncthreads();
#pragma unroll
    for (int off = 1; off < 256; off <<= 1) {
        int v = (tid >= off) ? lds[tid - off] : 0;
        __syncthreads();
        lds[tid] += v;
        __syncthreads();
    }
    int p = lds[tid] - tsum;  // exclusive prefix of this thread
    if (tid == 255) blockSums[blockIdx.x] = lds[255];
    if (base + 0 < n) offs[base + 0] = p;  p += c0;
    if (base + 1 < n) offs[base + 1] = p;  p += c1;
    if (base + 2 < n) offs[base + 2] = p;  p += c2;
    if (base + 3 < n) offs[base + 3] = p;
}

__global__ __launch_bounds__(256) void scan2_kernel(int* __restrict__ blockSums, int nb) {
    __shared__ int lds[256];
    const int tid = threadIdx.x;
    int v = (tid < nb) ? blockSums[tid] : 0;
    lds[tid] = v;
    __syncthreads();
#pragma unroll
    for (int off = 1; off < 256; off <<= 1) {
        int u = (tid >= off) ? lds[tid - off] : 0;
        __syncthreads();
        lds[tid] += u;
        __syncthreads();
    }
    if (tid < nb) blockSums[tid] = lds[tid] - v;  // exclusive
}

__global__ __launch_bounds__(256) void scan3_kernel(int* __restrict__ offs,
                                                    const int* __restrict__ blockSums,
                                                    int n, int total) {
    int idx = blockIdx.x * 256 + threadIdx.x;
    if (idx < n) offs[idx] += blockSums[idx >> 10];
    if (idx == 0) offs[n] = total;
}

__global__ __launch_bounds__(256) void fill_kernel(const int* __restrict__ row,
                                                   const int* __restrict__ col,
                                                   const float* __restrict__ val,
                                                   int* __restrict__ cursor,
                                                   int* __restrict__ colP,
                                                   float* __restrict__ valP, int nedges) {
    int e = blockIdx.x * 256 + threadIdx.x;
    if (e >= nedges) return;
    int p = atomicAdd(cursor + row[e], 1);
    colP[p] = col[e];
    valP[p] = val[e];
}

// ---------------- SpMM (CSR gather): one wave per destination row ----------------
// No float atomics; t (102MB) stays L2-resident so the 3.3GB gather is on-chip.
__global__ __launch_bounds__(256) void spmm_csr_kernel(const int* __restrict__ offs,
                                                       const int* __restrict__ colP,
                                                       const float* __restrict__ valP,
                                                       const float* __restrict__ t,
                                                       float* __restrict__ r, int nrows) {
    const int wid  = (blockIdx.x * blockDim.x + threadIdx.x) >> 5;
    const int lane = threadIdx.x & 31;
    if (wid >= nrows) return;
    const int start = __builtin_amdgcn_readfirstlane(offs[wid]);
    const int end   = __builtin_amdgcn_readfirstlane(offs[wid + 1]);
    float4 acc0 = {0.f, 0.f, 0.f, 0.f}, acc1 = {0.f, 0.f, 0.f, 0.f};
    for (int k0 = start; k0 < end; k0 += 32) {
        // batch-load up to 32 (col,val) pairs, one per lane; broadcast via shuffle
        int   ci = 0;
        float vv = 0.0f;
        const int kk = k0 + lane;
        if (kk < end) { ci = colP[kk]; vv = valP[kk]; }
        const int cnt = min(32, end - k0);
        for (int j = 0; j < cnt; j++) {
            int   cij = __shfl(ci, j, 32);
            float vj  = __shfl(vv, j, 32);
            const float4* src = (const float4*)(t + (size_t)cij * NHIDC);
            float4 x0 = src[lane * 2], x1 = src[lane * 2 + 1];
            acc0.x += vj * x0.x;  acc0.y += vj * x0.y;
            acc0.z += vj * x0.z;  acc0.w += vj * x0.w;
            acc1.x += vj * x1.x;  acc1.y += vj * x1.y;
            acc1.z += vj * x1.z;  acc1.w += vj * x1.w;
        }
    }
    float4* dst = (float4*)(r + (size_t)wid * NHIDC);
    dst[lane * 2]     = acc0;
    dst[lane * 2 + 1] = acc1;
}

// ---------------- Conv GEMM: h += act(r @ W^T + bias), W in LDS (f16) ----------------
// 32 output rows (2 strips) per wave: each LDS B fragment feeds 2 WMMAs.
template <bool RELU>
__global__ __launch_bounds__(256) void conv_gemm_kernel(const float* __restrict__ r,
                                                        const float* __restrict__ W,
                                                        const float* __restrict__ bias,
                                                        float* __restrict__ h, int npairs) {
    extern __shared__ _Float16 sW[];   // [256][LDS_CONV_STRIDE]
    const int tid = threadIdx.x;
    for (int it = 0; it < (NHIDC * NHIDC) / 256; it++) {
        int idx = tid + it * 256;
        sW[(idx >> 8) * LDS_CONV_STRIDE + (idx & 255)] = (_Float16)W[idx];
    }
    __syncthreads();

    const int wid = tid >> 5, lane = tid & 31;
    const int wstride = gridDim.x * 8;
    const int m = lane & 15;
    for (int pair = blockIdx.x * 8 + wid; pair < npairs; pair += wstride) {
        const int    m0    = pair * 32;
        const float* Arow0 = r + (size_t)(m0 + m) * NHIDC;
        const float* Arow1 = Arow0 + (size_t)16 * NHIDC;
        __builtin_prefetch(Arow0 + (size_t)wstride * 32 * NHIDC, 0, 0);  // next pair
        v16h af0[8], af1[8];
#pragma unroll
        for (int kc = 0; kc < 8; kc++) {
            af0[kc] = load_a_frag(Arow0, kc * 32, lane);
            af1[kc] = load_a_frag(Arow1, kc * 32, lane);
        }

#pragma unroll 1   // keep one acc pair + one B stream live -> stay under 256 VGPRs
        for (int nt = 0; nt < 16; nt++) {
            v8f acc0 = {}, acc1 = {};
            v16h bcur = load_b_frag(sW, LDS_CONV_STRIDE, nt * 16, 0, lane);
#pragma unroll
            for (int kc = 0; kc < 8; kc++) {
                v16h bnxt = bcur;                    // double-buffer: next B in flight
                if (kc < 7) bnxt = load_b_frag(sW, LDS_CONV_STRIDE, nt * 16, (kc + 1) * 32, lane);
                acc0 = wmma_f16(af0[kc], bcur, acc0);
                acc1 = wmma_f16(af1[kc], bcur, acc1);
                bcur = bnxt;
            }
            // C/D layout: lane<16 -> N=lane, M=j; lane>=16 -> N=lane-16, M=j+8
            const int   n     = nt * 16 + (lane & 15);
            const float bn    = bias[n];
            const int   mrow0 = (lane < 16) ? 0 : 8;
#pragma unroll
            for (int j = 0; j < 8; j++) {
                size_t off0 = (size_t)(m0 + mrow0 + j) * NHIDC + n;
                size_t off1 = off0 + (size_t)16 * NHIDC;
                float v0 = acc0[j] + bn;
                float v1 = acc1[j] + bn;
                if (RELU) { v0 = fmaxf(v0, 0.0f); v1 = fmaxf(v1, 0.0f); }
                h[off0] += v0;   // residual add (exclusive ownership, no atomics)
                h[off1] += v1;
            }
        }
    }
}

// ---------------- Input GEMM: h = relu(x @ in_W^T + b), K=500 padded to 512 ----------------
__global__ __launch_bounds__(256) void in_gemm_kernel(const float* __restrict__ x,
                                                      const float* __restrict__ W,
                                                      const float* __restrict__ bias,
                                                      float* __restrict__ h, int nstrips) {
    extern __shared__ _Float16 sW[];   // [256][LDS_IN_STRIDE], 266KB (WGP LDS = 320KB)
    const int tid = threadIdx.x;
    for (int it = 0; it < (NHIDC * 512) / 256; it++) {
        int idx = tid + it * 256;
        int n = idx >> 9, k = idx & 511;
        sW[n * LDS_IN_STRIDE + k] = (k < NFEAT) ? (_Float16)W[n * NFEAT + k] : (_Float16)0.0f;
    }
    __syncthreads();

    const int wid = tid >> 5, lane = tid & 31;
    const int wstride = gridDim.x * 8;
    const int m = lane & 15;
    for (int strip = blockIdx.x * 8 + wid; strip < nstrips; strip += wstride) {
        const float* Arow = x + (size_t)(strip * 16 + m) * NFEAT;
        v16h afrag[16];
#pragma unroll
        for (int kc = 0; kc < 15; kc++) afrag[kc] = load_a_frag(Arow, kc * 32, lane);
        afrag[15] = load_a_frag_guard(Arow, 15 * 32, lane, NFEAT);

#pragma unroll 1   // avoid interleaving 16 tiles -> stay under 256 VGPRs
        for (int nt = 0; nt < 16; nt++) {
            v8f acc = {};
            v16h bcur = load_b_frag(sW, LDS_IN_STRIDE, nt * 16, 0, lane);
#pragma unroll
            for (int kc = 0; kc < 16; kc++) {
                v16h bnxt = bcur;
                if (kc < 15) bnxt = load_b_frag(sW, LDS_IN_STRIDE, nt * 16, (kc + 1) * 32, lane);
                acc = wmma_f16(afrag[kc], bcur, acc);
                bcur = bnxt;
            }
            const int   n     = nt * 16 + (lane & 15);
            const float bn    = bias[n];
            const int   mrow0 = (lane < 16) ? 0 : 8;
#pragma unroll
            for (int j = 0; j < 8; j++) {
                size_t off = (size_t)(strip * 16 + mrow0 + j) * NHIDC + n;
                h[off] = fmaxf(acc[j] + bn, 0.0f);
            }
        }
    }
}

// ---------------- Output: z = LN(h) @ out_W^T + out_b (one wave per row) ----------------
__global__ __launch_bounds__(256) void out_kernel(const float* __restrict__ h,
                                                  const float* __restrict__ g,
                                                  const float* __restrict__ b,
                                                  const float* __restrict__ Wo,
                                                  const float* __restrict__ bo,
                                                  float* __restrict__ z, int nrows) {
    const int wid  = (blockIdx.x * blockDim.x + threadIdx.x) >> 5;
    const int lane = threadIdx.x & 31;
    if (wid >= nrows) return;
    const float4* hp = (const float4*)(h + (size_t)wid * NHIDC);
    float4 a0 = hp[lane * 2], a1 = hp[lane * 2 + 1];
    float s  = a0.x + a0.y + a0.z + a0.w + a1.x + a1.y + a1.z + a1.w;
    float s2 = a0.x*a0.x + a0.y*a0.y + a0.z*a0.z + a0.w*a0.w
             + a1.x*a1.x + a1.y*a1.y + a1.z*a1.z + a1.w*a1.w;
    s  = wave_sum32(s);
    s2 = wave_sum32(s2);
    const float mu  = s * (1.0f / NHIDC);
    const float var = s2 * (1.0f / NHIDC) - mu * mu;
    const float rs  = rsqrtf(var + EPSLN);
    const float4* gp = (const float4*)g;
    const float4* bp = (const float4*)b;
    float4 g0 = gp[lane * 2], g1 = gp[lane * 2 + 1];
    float4 b0 = bp[lane * 2], b1 = bp[lane * 2 + 1];
    float4 x0, x1;
    x0.x = (a0.x - mu) * rs * g0.x + b0.x;  x0.y = (a0.y - mu) * rs * g0.y + b0.y;
    x0.z = (a0.z - mu) * rs * g0.z + b0.z;  x0.w = (a0.w - mu) * rs * g0.w + b0.w;
    x1.x = (a1.x - mu) * rs * g1.x + b1.x;  x1.y = (a1.y - mu) * rs * g1.y + b1.y;
    x1.z = (a1.z - mu) * rs * g1.z + b1.z;  x1.w = (a1.w - mu) * rs * g1.w + b1.w;

    for (int c = 0; c < NCLS; c++) {
        const float4* wp = (const float4*)(Wo + (size_t)c * NHIDC) + lane * 2;
        float4 w0 = wp[0], w1 = wp[1];
        float p = x0.x*w0.x + x0.y*w0.y + x0.z*w0.z + x0.w*w0.w
                + x1.x*w1.x + x1.y*w1.y + x1.z*w1.z + x1.w*w1.w;
        p = wave_sum32(p);
        if (lane == 0) z[(size_t)wid * NCLS + c] = p + bo[c];
    }
}

// ---------------------------------------------------------------------------
extern "C" void kernel_launch(void* const* d_in, const int* in_sizes, int n_in,
                              void* d_out, int out_size, void* d_ws, size_t ws_size,
                              hipStream_t stream) {
    (void)in_sizes; (void)n_in; (void)out_size; (void)ws_size;
    const float* x     = (const float*)d_in[0];
    const int*   row   = (const int*)d_in[1];
    const int*   col   = (const int*)d_in[2];
    const float* val   = (const float*)d_in[3];
    const float* in_W  = (const float*)d_in[4];
    const float* in_b  = (const float*)d_in[5];
    const float* convW = (const float*)d_in[6];
    const float* convB = (const float*)d_in[7];
    const float* lnG   = (const float*)d_in[8];
    const float* lnB   = (const float*)d_in[9];
    const float* outG  = (const float*)d_in[10];
    const float* outB  = (const float*)d_in[11];
    const float* out_W = (const float*)d_in[12];
    const float* out_b = (const float*)d_in[13];
    float*       z     = (float*)d_out;

    // workspace layout
    float* h    = (float*)d_ws;                            // [N, 256]
    float* t    = h + (size_t)NNODE * NHIDC;               // [N, 256] (LN output)
    float* r    = t + (size_t)NNODE * NHIDC;               // [N, 256] (aggregation)
    float* valP = r + (size_t)NNODE * NHIDC;               // [E] reordered vals
    int*   colP = (int*)(valP + NEDGE);                    // [E] reordered cols
    int*   offs = colP + NEDGE;                            // [N+1] CSR offsets
    int*   curs = offs + (NNODE + 1);                      // [N] fill cursors
    int*   cnts = curs + NNODE;                            // [N] counts
    int*   bsum = cnts + NNODE;                            // [128] block sums

    const int nstrips   = NNODE / 16;                      // 6250
    const int npairs    = NNODE / 32;                      // 3125 (32-row conv tiles)
    const int rowBlocks = (NNODE + 7) / 8;                 // 8 waves/block, 1 row/wave
    const int edgeBlks  = (NEDGE + 255) / 256;             // thread/edge
    const int scanBlks  = (NNODE + 1023) / 1024;           // 98
    const int gemmBlks  = 256;                             // grid-stride; amortize W fill
    const size_t ldsIn   = (size_t)NHIDC * LDS_IN_STRIDE * sizeof(_Float16);
    const size_t ldsConv = (size_t)NHIDC * LDS_CONV_STRIDE * sizeof(_Float16);

    // ---- CSR build (once; reused by all 5 layers) ----
    hipMemsetAsync(cnts, 0, (size_t)NNODE * sizeof(int), stream);
    count_kernel<<<edgeBlks, 256, 0, stream>>>(row, cnts, NEDGE);
    scan1_kernel<<<scanBlks, 256, 0, stream>>>(cnts, offs, bsum, NNODE);
    scan2_kernel<<<1, 256, 0, stream>>>(bsum, scanBlks);
    scan3_kernel<<<(NNODE + 255) / 256, 256, 0, stream>>>(offs, bsum, NNODE, NEDGE);
    hipMemcpyAsync(curs, offs, (size_t)NNODE * sizeof(int), hipMemcpyDeviceToDevice, stream);
    fill_kernel<<<edgeBlks, 256, 0, stream>>>(row, col, val, curs, colP, valP, NEDGE);

    // ---- input layer ----
    in_gemm_kernel<<<gemmBlks, 256, ldsIn, stream>>>(x, in_W, in_b, h, nstrips);

    // ---- residual GCN layers ----
    for (int i = 0; i < NLAY; i++) {
        ln_kernel<<<rowBlocks, 256, 0, stream>>>(h, lnG + i * NHIDC, lnB + i * NHIDC, t, NNODE);
        spmm_csr_kernel<<<rowBlocks, 256, 0, stream>>>(offs, colP, valP, t, r, NNODE);
        const float* Wi = convW + (size_t)i * NHIDC * NHIDC;
        const float* Bi = convB + (size_t)i * NHIDC;
        if (i < NLAY - 1)
            conv_gemm_kernel<true><<<gemmBlks, 256, ldsConv, stream>>>(r, Wi, Bi, h, npairs);
        else
            conv_gemm_kernel<false><<<gemmBlks, 256, ldsConv, stream>>>(r, Wi, Bi, h, npairs);
    }

    out_kernel<<<rowBlocks, 256, 0, stream>>>(h, outG, outB, out_W, out_b, z, NNODE);
}
